// SelfAttention_20633022890642
// MI455X (gfx1250) — matrix operand order repaired and checked by reading, not compiled, not run on hardware
//
#include <hip/hip_runtime.h>
#include <math.h>

typedef __attribute__((ext_vector_type(16))) __bf16 v16bf;
typedef __attribute__((ext_vector_type(8)))  float  v8f;
typedef __attribute__((ext_vector_type(4)))  unsigned int v4u;
typedef __attribute__((ext_vector_type(8)))  int v8i;
typedef __attribute__((ext_vector_type(4)))  int v4i;

#define DEV __device__ __forceinline__

constexpr int Bx  = 2;
constexpr int Nx  = 2048;
constexpr int Dx  = 512;
constexpr int Hx  = 8;
constexpr int DKx = 64;
constexpr float NORMx = 0.125f; // 1/sqrt(64)

#if defined(__has_builtin)
#if __has_builtin(__builtin_amdgcn_tensor_load_to_lds) && \
    __has_builtin(__builtin_amdgcn_s_wait_tensorcnt)
#define USE_TDM 1
#endif
#endif

DEV v8f wmma_bf16(v16bf a, v16bf b, v8f c) {
  return __builtin_amdgcn_wmma_f32_16x16x32_bf16(
      false, a, false, b, (short)0, c, false, false);
}

DEV v16bf neg16(v16bf x) {
  v16bf r;
#pragma unroll
  for (int i = 0; i < 16; ++i) r[i] = (__bf16)(-(float)x[i]);
  return r;
}

// A fragment 16x32 bf16: elems 0..15 -> K = k0+half*16+i (contiguous)
DEV v16bf load_a_bf(const __bf16* rowptr, int k0, int half) {
  v16bf r;
  const __bf16* p0 = rowptr + k0 + half * 16;
  const __bf16* p1 = p0 + 8;
#pragma unroll
  for (int i = 0; i < 8; ++i) { r[i] = p0[i]; r[8 + i] = p1[i]; }
  return r;
}

DEV v16bf load16(const __bf16* p) {
  v16bf r;
#pragma unroll
  for (int i = 0; i < 16; ++i) r[i] = p[i];
  return r;
}

#ifdef USE_TDM
DEV unsigned lds_byte_off(const void* p) {
  return (unsigned)(unsigned long long)(const __attribute__((address_space(3))) void*)p;
}

// Issue one 2D TDM tile load: global (row-major, stride0 elems of 2B) -> LDS packed.
DEV void tdm_load_2d(unsigned lds_off, const void* gptr,
                     unsigned tile_d0, unsigned tile_d1,
                     unsigned tensor_d0, unsigned tensor_d1, unsigned stride0) {
  const unsigned long long ga = (unsigned long long)gptr;
  v4u g0;
  g0.x = 1u;                                         // count=1 (valid user D#)
  g0.y = lds_off;                                    // lds_addr (bytes)
  g0.z = (unsigned)(ga & 0xFFFFFFFFu);               // global_addr[31:0]
  g0.w = (unsigned)((ga >> 32) & 0x1FFFFFFu) | (2u << 30);  // addr[56:32] | type=2
  v8i g1;
  g1[0] = (int)(1u << 16);                           // data_size=1 (2 bytes)
  g1[1] = (int)((tensor_d0 & 0xFFFFu) << 16);        // tensor_dim0[15:0]
  g1[2] = (int)((tensor_d0 >> 16) | ((tensor_d1 & 0xFFFFu) << 16));
  g1[3] = (int)((tensor_d1 >> 16) | (tile_d0 << 16));  // tile_dim0
  g1[4] = (int)tile_d1;                              // tile_dim1 (tile_dim2=0)
  g1[5] = (int)stride0;                              // tensor_dim0_stride[31:0]
  g1[6] = 0;
  g1[7] = 0;
  const v4i z4 = {0, 0, 0, 0};                       // groups 2/3: zeros => 2D tensor
  const v8i z8 = {0, 0, 0, 0, 0, 0, 0, 0};
  __builtin_amdgcn_tensor_load_to_lds(g0, g1, z4, z4, z8, 0);
}
#endif

// --------------------------------------------------------------------------
// fp32 -> bf16 bulk convert (hoists all cvt out of the GEMM hot loops)
// --------------------------------------------------------------------------
__global__ __launch_bounds__(256)
void cvt_kernel(const float* __restrict__ in, __bf16* __restrict__ out, int n4) {
  const int i = blockIdx.x * 256 + threadIdx.x;
  if (i < n4) {
    const float4 v = ((const float4*)in)[i];
    out[4 * i + 0] = (__bf16)v.x;
    out[4 * i + 1] = (__bf16)v.y;
    out[4 * i + 2] = (__bf16)v.z;
    out[4 * i + 3] = (__bf16)v.w;
  }
}

// --------------------------------------------------------------------------
// Complex projection: Y = X @ W^T + b, all-bf16 operands, 32x32 tile per wave
// --------------------------------------------------------------------------
__global__ __launch_bounds__(32)
void proj_kernel(const __bf16* __restrict__ x_re, const __bf16* __restrict__ x_im,
                 const __bf16* __restrict__ W_re, const __bf16* __restrict__ W_im,
                 const float* __restrict__ b_re, const float* __restrict__ b_im,
                 __bf16* __restrict__ o_re, __bf16* __restrict__ o_im,
                 int transposed) {
  const int lane = threadIdx.x & 31;
  const int half = lane >> 4, lm = lane & 15;
  const int m0 = blockIdx.x * 32;
  const int j0 = blockIdx.y * 32;

  const __bf16* xr[2]; const __bf16* xi[2];
  const __bf16* wr[2]; const __bf16* wi[2];
#pragma unroll
  for (int t = 0; t < 2; ++t) {
    xr[t] = x_re + (size_t)(m0 + t * 16 + lm) * Dx;
    xi[t] = x_im + (size_t)(m0 + t * 16 + lm) * Dx;
    wr[t] = W_re + (size_t)(j0 + t * 16 + lm) * Dx;
    wi[t] = W_im + (size_t)(j0 + t * 16 + lm) * Dx;
  }

  v8f acr[2][2] = {}; v8f aci[2][2] = {};  // [m-tile][j-tile]

  for (int k = 0; k < Dx; k += 32) {
    v16bf ar[2], ai[2], an[2], br[2], bi[2];
#pragma unroll
    for (int mt = 0; mt < 2; ++mt) {
      ar[mt] = load_a_bf(xr[mt], k, half);
      ai[mt] = load_a_bf(xi[mt], k, half);
      an[mt] = neg16(ai[mt]);
    }
#pragma unroll
    for (int jt = 0; jt < 2; ++jt) {
      br[jt] = *(const v16bf*)(wr[jt] + k + half * 16);
      bi[jt] = *(const v16bf*)(wi[jt] + k + half * 16);
    }
#pragma unroll
    for (int mt = 0; mt < 2; ++mt)
#pragma unroll
      for (int jt = 0; jt < 2; ++jt) {
        acr[mt][jt] = wmma_bf16(ar[mt], br[jt], acr[mt][jt]);
        acr[mt][jt] = wmma_bf16(an[mt], bi[jt], acr[mt][jt]);  // - xi*Wi
        aci[mt][jt] = wmma_bf16(ar[mt], bi[jt], aci[mt][jt]);
        aci[mt][jt] = wmma_bf16(ai[mt], br[jt], aci[mt][jt]);
      }
  }

#pragma unroll
  for (int jt = 0; jt < 2; ++jt) {
    const int j = j0 + jt * 16 + lm;      // C-layout column
    const float br = b_re[j], bi = b_im[j];
#pragma unroll
    for (int mt = 0; mt < 2; ++mt) {
      if (!transposed) {
#pragma unroll
        for (int r = 0; r < 8; ++r) {
          const int gm = m0 + mt * 16 + 8 * half + r;  // C-layout row
          o_re[(size_t)gm * Dx + j] = (__bf16)(acr[mt][jt][r] + br);
          o_im[(size_t)gm * Dx + j] = (__bf16)(aci[mt][jt][r] + bi);
        }
      } else {
        const int hh = j >> 6, dl = j & 63;
#pragma unroll
        for (int r = 0; r < 8; ++r) {
          const int gm = m0 + mt * 16 + 8 * half + r;
          const int bb = gm >> 11, n = gm & (Nx - 1);
          const size_t idx = (((size_t)(bb * Hx + hh)) * DKx + dl) * Nx + n;
          o_re[idx] = (__bf16)(acr[mt][jt][r] + br);
          o_im[idx] = (__bf16)(aci[mt][jt][r] + bi);
        }
      }
    }
  }
}

// --------------------------------------------------------------------------
// Flash attention. 4 waves / block share one head's K/V chunk through LDS
// (TDM async tile DMA when available), each wave owns a 16-row query tile.
// --------------------------------------------------------------------------
__global__ __launch_bounds__(128)
void attn_kernel(const __bf16* __restrict__ q_re, const __bf16* __restrict__ q_im,
                 const __bf16* __restrict__ k_re, const __bf16* __restrict__ k_im,
                 const __bf16* __restrict__ vT_re, const __bf16* __restrict__ vT_im,
                 float* __restrict__ out) {
  __shared__ __bf16 Kre[32][64];   // [key][d]   (row-major, d contiguous)
  __shared__ __bf16 Kim[32][64];
  __shared__ __bf16 Vre[64][32];   // [d][key]   (keys contiguous)
  __shared__ __bf16 Vim[64][32];
  __shared__ __bf16 Plds[4][16][32];

  const int tid  = threadIdx.x;
  const int wave = tid >> 5, lane = tid & 31;
  const int half = lane >> 4, lm = lane & 15;
  const int q0 = blockIdx.x * 64 + wave * 16;
  const int hh = blockIdx.y;
  const int b  = blockIdx.z;

  const size_t qbase = ((size_t)(b * Nx) + q0 + lm) * Dx + hh * DKx;
  const v16bf qre0 = load_a_bf(q_re + qbase, 0, half);
  const v16bf qre1 = load_a_bf(q_re + qbase, 32, half);
  const v16bf qim0 = load_a_bf(q_im + qbase, 0, half);
  const v16bf qim1 = load_a_bf(q_im + qbase, 32, half);
  const v16bf nqi0 = neg16(qim0);
  const v16bf nqi1 = neg16(qim1);

  v8f accr[4] = {}; v8f acci[4] = {};
  float mrow[8], lrow[8];
#pragma unroll
  for (int r = 0; r < 8; ++r) { mrow[r] = -1e30f; lrow[r] = 0.0f; }

  const size_t vbase = ((size_t)(b * Hx + hh)) * DKx * Nx;

  for (int m0 = 0; m0 < Nx; m0 += 32) {
    // ---- stage K/V chunk (32 keys) into LDS, shared by all 4 waves ----
#ifdef USE_TDM
    if (wave == 0) {
      const __bf16* gkr = k_re + ((size_t)(b * Nx) + m0) * Dx + hh * DKx;
      const __bf16* gki = k_im + ((size_t)(b * Nx) + m0) * Dx + hh * DKx;
      const __bf16* gvr = vT_re + vbase + m0;
      const __bf16* gvi = vT_im + vbase + m0;
      tdm_load_2d(lds_byte_off(&Kre[0][0]), gkr, DKx, 32, Dx, (unsigned)(Bx * Nx), Dx);
      tdm_load_2d(lds_byte_off(&Kim[0][0]), gki, DKx, 32, Dx, (unsigned)(Bx * Nx), Dx);
      tdm_load_2d(lds_byte_off(&Vre[0][0]), gvr, 32, DKx, Nx, (unsigned)(Bx * Hx * DKx), Nx);
      tdm_load_2d(lds_byte_off(&Vim[0][0]), gvi, 32, DKx, Nx, (unsigned)(Bx * Hx * DKx), Nx);
      __builtin_amdgcn_s_wait_tensorcnt(0);
    }
#else
    {
      const int key = tid >> 2, seg = (tid & 3) * 16;        // K: 32x64
      const __bf16* gr = k_re + ((size_t)(b * Nx) + m0 + key) * Dx + hh * DKx + seg;
      const __bf16* gi = k_im + ((size_t)(b * Nx) + m0 + key) * Dx + hh * DKx + seg;
#pragma unroll
      for (int i = 0; i < 16; ++i) { Kre[key][seg + i] = gr[i]; Kim[key][seg + i] = gi[i]; }
      const int dr = tid >> 1, vs = (tid & 1) * 16;          // V: 64x32
      const __bf16* vr = vT_re + vbase + (size_t)dr * Nx + m0 + vs;
      const __bf16* vi = vT_im + vbase + (size_t)dr * Nx + m0 + vs;
#pragma unroll
      for (int i = 0; i < 16; ++i) { Vre[dr][vs + i] = vr[i]; Vim[dr][vs + i] = vi[i]; }
    }
#endif
    __syncthreads();

    // ---- scores for two 16-key tiles ----
    float aval[2][8];
#pragma unroll
    for (int t = 0; t < 2; ++t) {
      const int key = t * 16 + lm;
      const v16bf kr0 = load16(&Kre[key][half * 16]);
      const v16bf kr1 = load16(&Kre[key][32 + half * 16]);
      const v16bf ki0 = load16(&Kim[key][half * 16]);
      const v16bf ki1 = load16(&Kim[key][32 + half * 16]);
      v8f sre = {}; v8f sim = {};
      sre = wmma_bf16(qre0, kr0, sre);
      sre = wmma_bf16(qre1, kr1, sre);
      sre = wmma_bf16(nqi0, ki0, sre);
      sre = wmma_bf16(nqi1, ki1, sre);
      sim = wmma_bf16(qre0, ki0, sim);
      sim = wmma_bf16(qre1, ki1, sim);
      sim = wmma_bf16(qim0, kr0, sim);
      sim = wmma_bf16(qim1, kr1, sim);
#pragma unroll
      for (int r = 0; r < 8; ++r)
        aval[t][r] = sqrtf(sre[r] * sre[r] + sim[r] * sim[r]) * NORMx;
    }

    // ---- online softmax (row stats via shfl-xor within each 16-lane half) ----
    float p0[8], p1[8], scl[8];
#pragma unroll
    for (int r = 0; r < 8; ++r) {
      float tmax = fmaxf(aval[0][r], aval[1][r]);
#pragma unroll
      for (int msk = 8; msk >= 1; msk >>= 1)
        tmax = fmaxf(tmax, __shfl_xor(tmax, msk, 32));
      const float mnew = fmaxf(mrow[r], tmax);
      scl[r] = __expf(mrow[r] - mnew);
      mrow[r] = mnew;
      p0[r] = __expf(aval[0][r] - mnew);
      p1[r] = __expf(aval[1][r] - mnew);
      float ps = p0[r] + p1[r];
#pragma unroll
      for (int msk = 8; msk >= 1; msk >>= 1)
        ps += __shfl_xor(ps, msk, 32);
      lrow[r] = lrow[r] * scl[r] + ps;
    }
#pragma unroll
    for (int dt = 0; dt < 4; ++dt)
#pragma unroll
      for (int r = 0; r < 8; ++r) { accr[dt][r] *= scl[r]; acci[dt][r] *= scl[r]; }

    // ---- P: C layout -> A layout via LDS transpose (per-wave region) ----
#pragma unroll
    for (int r = 0; r < 8; ++r) {
      Plds[wave][8 * half + r][lm]      = (__bf16)p0[r];
      Plds[wave][8 * half + r][16 + lm] = (__bf16)p1[r];
    }
    __syncthreads();
    v16bf Pfrag;
#pragma unroll
    for (int i = 0; i < 8; ++i) {
      Pfrag[i]     = Plds[wave][lm][half * 16 + i];
      Pfrag[8 + i] = Plds[wave][lm][half * 16 + 8 + i];
    }

    // ---- P @ V (K = 32 keys) from LDS ----
#pragma unroll
    for (int dt = 0; dt < 4; ++dt) {
      const v16bf vr = load16(&Vre[dt * 16 + lm][half * 16]);
      const v16bf vi = load16(&Vim[dt * 16 + lm][half * 16]);
      accr[dt] = wmma_bf16(Pfrag, vr, accr[dt]);
      acci[dt] = wmma_bf16(Pfrag, vi, acci[dt]);
    }
    __syncthreads();  // protect K/V/P tiles before next chunk's staging
  }

  // ---- epilogue: normalize and write [B, N, D, 2] fp32 ----
#pragma unroll
  for (int dt = 0; dt < 4; ++dt) {
#pragma unroll
    for (int r = 0; r < 8; ++r) {
      const int n = q0 + 8 * half + r;
      const int d = hh * DKx + dt * 16 + lm;
      const float inv = 1.0f / lrow[r];
      const size_t o = (((size_t)b * Nx + n) * Dx + d) * 2;
      out[o]     = accr[dt][r] * inv;
      out[o + 1] = acci[dt][r] * inv;
    }
  }
}

extern "C" void kernel_launch(void* const* d_in, const int* in_sizes, int n_in,
                              void* d_out, int out_size, void* d_ws, size_t ws_size,
                              hipStream_t stream) {
  (void)in_sizes; (void)n_in; (void)out_size; (void)ws_size;
  const float* x_re  = (const float*)d_in[0];
  const float* x_im  = (const float*)d_in[1];
  const float* Wq_re = (const float*)d_in[2];
  const float* Wq_im = (const float*)d_in[3];
  const float* bq_re = (const float*)d_in[4];
  const float* bq_im = (const float*)d_in[5];
  const float* Wk_re = (const float*)d_in[6];
  const float* Wk_im = (const float*)d_in[7];
  const float* bk_re = (const float*)d_in[8];
  const float* bk_im = (const float*)d_in[9];
  const float* Wv_re = (const float*)d_in[10];
  const float* Wv_im = (const float*)d_in[11];
  const float* bv_re = (const float*)d_in[12];
  const float* bv_im = (const float*)d_in[13];

  const size_t E  = (size_t)Bx * Nx * Dx;   // 2,097,152
  const size_t EW = (size_t)Dx * Dx;        //   262,144
  __bf16* ws  = (__bf16*)d_ws;
  __bf16* qre = ws;
  __bf16* qim = ws + E;
  __bf16* kre = ws + 2 * E;
  __bf16* kim = ws + 3 * E;
  __bf16* vtr = ws + 4 * E;
  __bf16* vti = ws + 5 * E;
  __bf16* xbr = ws + 6 * E;
  __bf16* xbi = ws + 7 * E;
  __bf16* wb  = ws + 8 * E;                 // 6 weight matrices, bf16
  __bf16* wqr = wb;          __bf16* wqi = wb + EW;
  __bf16* wkr = wb + 2 * EW; __bf16* wki = wb + 3 * EW;
  __bf16* wvr = wb + 4 * EW; __bf16* wvi = wb + 5 * EW;

  const int nx4 = (int)(E / 4), nw4 = (int)(EW / 4);
  cvt_kernel<<<dim3((nx4 + 255) / 256), 256, 0, stream>>>(x_re, xbr, nx4);
  cvt_kernel<<<dim3((nx4 + 255) / 256), 256, 0, stream>>>(x_im, xbi, nx4);
  cvt_kernel<<<dim3((nw4 + 255) / 256), 256, 0, stream>>>(Wq_re, wqr, nw4);
  cvt_kernel<<<dim3((nw4 + 255) / 256), 256, 0, stream>>>(Wq_im, wqi, nw4);
  cvt_kernel<<<dim3((nw4 + 255) / 256), 256, 0, stream>>>(Wk_re, wkr, nw4);
  cvt_kernel<<<dim3((nw4 + 255) / 256), 256, 0, stream>>>(Wk_im, wki, nw4);
  cvt_kernel<<<dim3((nw4 + 255) / 256), 256, 0, stream>>>(Wv_re, wvr, nw4);
  cvt_kernel<<<dim3((nw4 + 255) / 256), 256, 0, stream>>>(Wv_im, wvi, nw4);

  dim3 pg(Bx * Nx / 32, Dx / 32), blk(32);
  proj_kernel<<<pg, blk, 0, stream>>>(xbr, xbi, wqr, wqi, bq_re, bq_im, qre, qim, 0);
  proj_kernel<<<pg, blk, 0, stream>>>(xbr, xbi, wkr, wki, bk_re, bk_im, kre, kim, 0);
  proj_kernel<<<pg, blk, 0, stream>>>(xbr, xbi, wvr, wvi, bv_re, bv_im, vtr, vti, 1);

  attn_kernel<<<dim3(Nx / 64, Hx, Bx), 128, 0, stream>>>(qre, qim, kre, kim, vtr, vti,
                                                         (float*)d_out);
}
